// AudioModel_63393717289128
// MI455X (gfx1250) — matrix operand order repaired and checked by reading, not compiled, not run on hardware
//
#include <hip/hip_runtime.h>
#include <hip/hip_bf16.h>

// LSTM (B=8192, T=100, in=40, H=64) + MLP head, fully fused.
// Strategy: weights resident in VGPRs as WMMA A-fragments (f16), one wave per
// gate type (i/f/g/o), 16-batch tile per workgroup, h/c state on-chip.

#define N_MFCC   40
#define HIDDEN   64
#define T_STEPS  100
#define BATCH    8192
#define BTILE    16              // batch rows per workgroup (WMMA N dim)
#define NWAVES   4               // one wave per gate (i,f,g,o)
#define WG_THREADS (NWAVES * 32)
#define GROW     20              // padded row stride (floats) for gates_lds
#define HROW     68              // padded row stride (halves) for h_lds

typedef __attribute__((ext_vector_type(16))) _Float16 v16h;
typedef __attribute__((ext_vector_type(8)))  float    v8f;

__device__ __forceinline__ float fast_sigmoid(float x) {
    return 1.0f / (1.0f + __expf(-x));
}
__device__ __forceinline__ float fast_tanh(float x) {
    return 2.0f / (1.0f + __expf(-2.0f * x)) - 1.0f;
}

__global__ __launch_bounds__(WG_THREADS, 1)
void lstm_wmma_kernel(const float* __restrict__ feats,
                      const float* __restrict__ W_ih,
                      const float* __restrict__ W_hh,
                      const float* __restrict__ b_ih,
                      const float* __restrict__ b_hh,
                      const float* __restrict__ W1,
                      const float* __restrict__ b1,
                      const float* __restrict__ W2,
                      const float* __restrict__ b2,
                      float* __restrict__ out)
{
    __shared__ float    x_lds[BTILE][N_MFCC];      // staged x_t (fp32)
    __shared__ _Float16 h_lds[BTILE][HROW];        // hidden state (f16), padded
    __shared__ float    gates_lds[NWAVES][64 * GROW]; // raw gate outputs (f32)
    __shared__ float    bias_lds[4 * HIDDEN];      // b_ih + b_hh combined

    const int tid        = threadIdx.x;
    const int wave       = tid >> 5;               // gate type: 0=i,1=f,2=g,3=o
    const int lane       = tid & 31;
    const int ln         = lane & 15;              // M-row / N-col within tile
    const int grp        = lane >> 4;              // lane half (fragment layout)
    const int batch_base = blockIdx.x * BTILE;
    const int gate_base  = wave * 64;

    // Combined bias (reference adds b_ih + b_hh once).
    for (int i = tid; i < 4 * HIDDEN; i += WG_THREADS)
        bias_lds[i] = b_ih[i] + b_hh[i];
    // h0 = 0
    for (int i = tid; i < BTILE * HROW; i += WG_THREADS)
        (&h_lds[0][0])[i] = (_Float16)0.0f;

    // ---- Build resident A fragments (weights, f16), per gfx1250 A layout:
    //   lane l holds row M = (l&15); element e -> K = 16*(e>>3) + 8*(l>>4) + (e&7)
    // K packing: [0..39]=input feats, [40..63]=zero pad, [64..127]=hidden.
    v16h afrag[4][4];
#pragma unroll
    for (int mt = 0; mt < 4; ++mt) {
        const int row = gate_base + mt * 16 + ln;  // gate output dim
#pragma unroll
        for (int kt = 0; kt < 4; ++kt) {
#pragma unroll
            for (int e = 0; e < 16; ++e) {
                const int K = kt * 32 + 16 * (e >> 3) + 8 * grp + (e & 7);
                float v;
                if (K < 64) v = (K < N_MFCC) ? W_ih[row * N_MFCC + K] : 0.0f;
                else        v = W_hh[row * HIDDEN + (K - 64)];
                afrag[mt][kt][e] = (_Float16)v;
            }
        }
    }

    // Cell state: fixed (m,n) assignment per thread across all timesteps.
    float cstate[8];
#pragma unroll
    for (int p = 0; p < 8; ++p) cstate[p] = 0.0f;

    __syncthreads();

#pragma unroll 1
    for (int t = 0; t < T_STEPS; ++t) {
        // ---- Stage x_t into LDS once per workgroup (coalesced-ish).
        for (int i = tid; i < BTILE * N_MFCC; i += WG_THREADS) {
            const int n = i / N_MFCC;
            const int k = i - n * N_MFCC;
            x_lds[n][k] =
                feats[((size_t)(batch_base + n) * T_STEPS + t) * N_MFCC + k];
        }
        __syncthreads();

        // ---- Build B fragments: lane l holds col N = (l&15);
        //      element e -> K = kt*32 + 16*(e>>3) + 8*(l>>4) + (e&7).
        v16h bfrag[4];
#pragma unroll
        for (int kt = 0; kt < 4; ++kt) {
#pragma unroll
            for (int e = 0; e < 16; ++e) {
                const int K = kt * 32 + 16 * (e >> 3) + 8 * grp + (e & 7);
                _Float16 v;
                if (K < 64) v = (K < N_MFCC) ? (_Float16)x_lds[ln][K]
                                             : (_Float16)0.0f;
                else        v = h_lds[ln][K - 64];
                bfrag[kt][e] = v;
            }
        }

        // ---- gates[64,16] = W_gate[64,128] x xh[128,16]  (16 WMMAs/wave)
#pragma unroll
        for (int mt = 0; mt < 4; ++mt) {
            v8f acc = {0.f, 0.f, 0.f, 0.f, 0.f, 0.f, 0.f, 0.f};
#pragma unroll
            for (int kt = 0; kt < 4; ++kt) {
                acc = __builtin_amdgcn_wmma_f32_16x16x32_f16(
                    false, afrag[mt][kt], false, bfrag[kt],
                    (short)0, acc, false, false);
            }
            // C/D layout: lane l, vgpr r -> M = r + 8*(l>>4), N = l&15
#pragma unroll
            for (int r = 0; r < 8; ++r) {
                const int m = mt * 16 + r + 8 * grp;
                gates_lds[wave][m * GROW + ln] = acc[r];
            }
        }
        __syncthreads();

        // ---- Elementwise LSTM cell: 1024 (m,n) pairs over 128 threads.
#pragma unroll
        for (int p = 0; p < 8; ++p) {
            const int idx = p * WG_THREADS + tid;
            const int m = idx >> 4;    // hidden dim 0..63
            const int n = idx & 15;    // batch col
            const float ig = fast_sigmoid(gates_lds[0][m * GROW + n] + bias_lds[0 * 64 + m]);
            const float fg = fast_sigmoid(gates_lds[1][m * GROW + n] + bias_lds[1 * 64 + m]);
            const float gg = fast_tanh  (gates_lds[2][m * GROW + n] + bias_lds[2 * 64 + m]);
            const float og = fast_sigmoid(gates_lds[3][m * GROW + n] + bias_lds[3 * 64 + m]);
            const float c  = fg * cstate[p] + ig * gg;
            cstate[p] = c;
            h_lds[n][m] = (_Float16)(og * fast_tanh(c));
        }
        __syncthreads();
    }

    // ---- Classifier head: relu(h @ W1^T + b1) @ W2^T + b2, one row/thread.
    if (tid < BTILE) {
        const int n = tid;
        float score = b2[0];
#pragma unroll 1
        for (int j = 0; j < 32; ++j) {
            float acc = b1[j];
#pragma unroll
            for (int k = 0; k < HIDDEN; ++k)
                acc += (float)h_lds[n][k] * W1[j * HIDDEN + k];
            score += fmaxf(acc, 0.0f) * W2[j];
        }
        out[batch_base + n] = score;
    }
}

extern "C" void kernel_launch(void* const* d_in, const int* in_sizes, int n_in,
                              void* d_out, int out_size, void* d_ws, size_t ws_size,
                              hipStream_t stream) {
    (void)in_sizes; (void)n_in; (void)d_ws; (void)ws_size; (void)out_size;
    const float* feats = (const float*)d_in[0];
    const float* W_ih  = (const float*)d_in[1];
    const float* W_hh  = (const float*)d_in[2];
    const float* b_ih  = (const float*)d_in[3];
    const float* b_hh  = (const float*)d_in[4];
    const float* W1    = (const float*)d_in[5];
    const float* b1    = (const float*)d_in[6];
    const float* W2    = (const float*)d_in[7];
    const float* b2    = (const float*)d_in[8];
    float* out = (float*)d_out;

    dim3 grid(BATCH / BTILE);   // 512 workgroups
    dim3 block(WG_THREADS);     // 128 threads = 4 waves (wave32)
    hipLaunchKernelGGL(lstm_wmma_kernel, grid, block, 0, stream,
                       feats, W_ih, W_hh, b_ih, b_hh, W1, b1, W2, b2, out);
}
